// SimpleBigram_19971597926680
// MI455X (gfx1250) — matrix operand order repaired and checked by reading, not compiled, not run on hardware
//
#include <hip/hip_runtime.h>

// ---------------- problem constants ----------------
#define VOCAB 50257
#define SEQ   256
#define DEMB  256
#define DHID  512
#define BATCH 16
#define G4    2048           // 4*DHID
#define ROWS  4096           // BATCH*SEQ
#define VPAD  50688          // 792 * 64, padded vocab for tail-free B loads

typedef __attribute__((ext_vector_type(16))) __bf16 v16bf;
typedef __attribute__((ext_vector_type(8)))  __bf16 v8bf;
typedef __attribute__((ext_vector_type(8)))  float  v8f;

// --------- CDNA5 async global->LDS path (probe via __has_builtin, safe fallback) ---------
#if defined(__has_builtin)
#if __has_builtin(__builtin_amdgcn_global_load_async_to_lds_b128) && \
    __has_builtin(__builtin_amdgcn_s_wait_asynccnt)
#define USE_ASYNC_LDS 1
#endif
#endif

// exact builtin parameter types (from clang diagnostic): AS(1)/AS(3) pointers to int32x4
typedef int b128i __attribute__((vector_size(16)));
typedef __attribute__((address_space(1))) b128i* gas_b128p;
typedef __attribute__((address_space(3))) b128i* las_b128p;

__device__ __forceinline__ void copy16_g2l(__bf16* lds_dst, const __bf16* gsrc) {
#ifdef USE_ASYNC_LDS
  __builtin_amdgcn_global_load_async_to_lds_b128(
      (gas_b128p)gsrc, (las_b128p)lds_dst, 0, 0);
#else
  *(v8bf*)lds_dst = *(const v8bf*)gsrc;   // synchronous fallback
#endif
}

__device__ __forceinline__ void async_copies_done() {
#ifdef USE_ASYNC_LDS
  __builtin_amdgcn_s_wait_asynccnt(0);
#endif
}

// ---------------- WMMA fragment loaders (wave32, 16x16x32 bf16) ----------------
// A 16x32 layout: lane L holds row M=L%16; 16 contiguous K [off..off+15],
// off = k0 + 16*(L/16).
__device__ __forceinline__ v16bf load_a_bf16(const __bf16* __restrict__ row, int k0, int lane) {
  int off = k0 + ((lane >> 4) << 4);
  v8bf lo = *(const v8bf*)(row + off);
  v8bf hi = *(const v8bf*)(row + off + 8);
  return __builtin_shufflevector(lo, hi, 0,1,2,3,4,5,6,7,8,9,10,11,12,13,14,15);
}

__device__ __forceinline__ v16bf load_a_f32(const float* __restrict__ row, int k0, int lane) {
  int off = k0 + ((lane >> 4) << 4);
  v16bf a;
#pragma unroll
  for (int e = 0; e < 8; ++e) {
    a[e]     = (__bf16)row[off + e];
    a[e + 8] = (__bf16)row[off + 8 + e];
  }
  return a;
}

// B 32x16 layout: lane L holds col N=L%16; 16 contiguous K (32B) at k0 + 16*(L/16).
// `colrow` = row n of the pre-transposed (K-major) weight matrix.
__device__ __forceinline__ v16bf load_b_bf16(const __bf16* __restrict__ colrow, int k0, int lane) {
  return *(const v16bf*)(colrow + k0 + ((lane >> 4) << 4));
}

__device__ __forceinline__ v8f wmma_bf16(v16bf a, v16bf b, v8f c) {
  return __builtin_amdgcn_wmma_f32_16x16x32_bf16(false, a, false, b, (short)0, c, false, false);
}

// ---------------- kernel 0: fp32 [K][N] -> bf16 transposed [Npad][K], zero padded ----------------
__global__ void cvt_transpose_kernel(const float* __restrict__ src, __bf16* __restrict__ dst,
                                     int K, int N, int Npad) {
  long id = (long)blockIdx.x * blockDim.x + threadIdx.x;
  long total = (long)Npad * K;
  if (id >= total) return;
  int n  = (int)(id / K);
  int kk = (int)(id - (long)n * K);
  float v = (n < N) ? src[(long)kk * N + n] : 0.0f;
  dst[(long)n * K + kk] = (__bf16)v;
}

__global__ void init_bar_kernel(unsigned int* bar) { *bar = 0u; }

// ---------------- kernel 1: xz[s*16+b][g] = E[idx[b][s]] @ Wi  (WMMA bf16) ----------------
// grid (4, 256), block 256 (8 waves). Wave w: 16 rows x 64 cols.
__global__ void __launch_bounds__(256) xz_gemm_kernel(const int* __restrict__ idx,
                                                      const float* __restrict__ E,
                                                      const __bf16* __restrict__ WiT,
                                                      float* __restrict__ xz) {
  int lane = threadIdx.x & 31;
  int wave = threadIdx.x >> 5;
  int s  = blockIdx.y;                       // timestep = row-tile
  int c0 = blockIdx.x * 512 + wave * 64;     // column base for this wave
  int b  = lane & 15;                        // batch row this lane supplies
  int token = idx[b * SEQ + s];
  const float* arow = E + (long)token * DEMB;

  v8f acc[4] = {};
#pragma unroll
  for (int k0 = 0; k0 < DEMB; k0 += 32) {
    v16bf a = load_a_f32(arow, k0, lane);
#pragma unroll
    for (int t = 0; t < 4; ++t) {
      const __bf16* brow = WiT + (long)(c0 + t * 16 + (lane & 15)) * DEMB;
      v16bf bf = load_b_bf16(brow, k0, lane);
      acc[t] = wmma_bf16(a, bf, acc[t]);
    }
  }
  int mbase = 8 * (lane >> 4);
#pragma unroll
  for (int t = 0; t < 4; ++t) {
    int col = c0 + t * 16 + (lane & 15);
#pragma unroll
    for (int i = 0; i < 8; ++i) {
      int brw = mbase + i;                               // batch index
      xz[((long)s * BATCH + brw) * G4 + col] = acc[t][i];
    }
  }
}

// ---------------- kernel 2: LSTM scan, 16 WGs (one cluster), hidden sharded 32/WG ----------------
// Each wave's 16 WhT columns (16x512 bf16 = 128 VGPRs) are preloaded into registers once;
// the 256-step time loop then only streams h (16 KB) + does register-resident WMMAs.
__global__ void __launch_bounds__(256, 1) lstm_scan_kernel(const float* __restrict__ xz,
                                                           const __bf16* __restrict__ WhT,
                                                           const float* __restrict__ bh,
                                                           __bf16* __restrict__ h0buf,
                                                           __bf16* __restrict__ h1buf,
                                                           __bf16* __restrict__ hs,
                                                           unsigned int* __restrict__ bar) {
  __shared__ float z_lds[BATCH][4][32];  // h@Wh partials: [b][gate][local hidden]
  __shared__ float c_lds[BATCH][32];     // cell state for owned hidden slice

  const int wg   = blockIdx.x;           // 0..15 : owns hidden [wg*32, wg*32+32)
  const int lane = threadIdx.x & 31;
  const int wave = threadIdx.x >> 5;     // 8 waves
  const int g    = wave >> 1;            // gate 0..3 (i,f,g,o)
  const int t    = wave & 1;             // 16-col half of the 32-hidden slice

  // init: h0 slice = 0, c = 0
  for (int id = threadIdx.x; id < BATCH * 32; id += 256) {
    int b = id >> 5, jj = id & 31;
    h0buf[b * DHID + wg * 32 + jj] = (__bf16)0.0f;
    c_lds[b][jj] = 0.0f;
  }

  // Preload this wave's entire B operand (its 16 gate columns, all K) into registers.
  const int ncol = g * DHID + wg * 32 + t * 16 + (lane & 15);   // global gate column
  const __bf16* brow = WhT + (long)ncol * DHID;
  v16bf bfr[16];
#pragma unroll
  for (int kk = 0; kk < 16; ++kk) bfr[kk] = load_b_bf16(brow, kk * 32, lane);

  // --- barrier 0: all slices of h0 initialized, visible at device scope ---
  __threadfence();
  __syncthreads();
  if (threadIdx.x == 0) {
    atomicAdd(bar, 1u);
    while (*(volatile unsigned int*)bar < 16u) __builtin_amdgcn_s_sleep(1);
  }
  if (threadIdx.x < 32) __builtin_amdgcn_s_cluster_barrier();   // ISA NOP if not clustered
  __syncthreads();
  __threadfence();

  for (int s = 0; s < SEQ; ++s) {
    const __bf16* h_in  = (s & 1) ? h1buf : h0buf;
    __bf16*       h_out = (s & 1) ? h0buf : h1buf;

    // z_partial = h @ Wh for this wave's 16 columns (M=16, K=512), B in registers
    const __bf16* arow = h_in + (long)(lane & 15) * DHID;
    v8f acc = {};
#pragma unroll
    for (int kk = 0; kk < 16; ++kk) {
      v16bf a = load_a_bf16(arow, kk * 32, lane);
      acc = wmma_bf16(a, bfr[kk], acc);
    }
#pragma unroll
    for (int i = 0; i < 8; ++i) {
      int M = i + 8 * (lane >> 4);
      z_lds[M][g][t * 16 + (lane & 15)] = acc[i];
    }
    __syncthreads();

    // cell update: 512 (b,jj) pairs over 256 threads
#pragma unroll
    for (int p = 0; p < 2; ++p) {
      int id = threadIdx.x + p * 256;
      int b = id >> 5, jj = id & 31;
      int j = wg * 32 + jj;
      const float* xzrow = xz + ((long)s * BATCH + b) * G4;
      float zi = z_lds[b][0][jj] + xzrow[j]            + bh[j];
      float zf = z_lds[b][1][jj] + xzrow[DHID + j]     + bh[DHID + j];
      float zg = z_lds[b][2][jj] + xzrow[2 * DHID + j] + bh[2 * DHID + j];
      float zo = z_lds[b][3][jj] + xzrow[3 * DHID + j] + bh[3 * DHID + j];
      float iv = 1.0f / (1.0f + __expf(-zi));
      float fv = 1.0f / (1.0f + __expf(-zf));
      float gv = tanhf(zg);
      float ov = 1.0f / (1.0f + __expf(-zo));
      float cc = fv * c_lds[b][jj] + iv * gv;
      c_lds[b][jj] = cc;
      __bf16 hb = (__bf16)(ov * tanhf(cc));
      h_out[b * DHID + j] = hb;                                  // ping-pong state
      hs[((long)b * SEQ + s) * DHID + j] = hb;                   // [B,S,H] for the head
    }

    // --- per-step barrier: new h visible to all 16 WGs ---
    __threadfence();
    __syncthreads();
    if (threadIdx.x == 0) {
      atomicAdd(bar, 1u);
      unsigned int target = 16u * (unsigned int)(s + 2);
      while (*(volatile unsigned int*)bar < target) __builtin_amdgcn_s_sleep(1);
    }
    if (threadIdx.x < 32) __builtin_amdgcn_s_cluster_barrier();
    __syncthreads();
    __threadfence();
  }
}

// ---------------- kernel 3: logits = hs @ Wo + bo  (dominant GEMM, WMMA bf16) ----------------
// Block tile 128 rows x 64 cols, 8 waves (one 16-row M-tile each) SHARE the B tile.
// B staged in LDS (double-buffered 4 KB stages) via async global->LDS copies.
// grid (VPAD/64 = 792, ROWS/128 = 32), block 256.
__global__ void __launch_bounds__(256) logits_gemm_kernel(const __bf16* __restrict__ hs,
                                                          const __bf16* __restrict__ WoT,
                                                          const float* __restrict__ bo,
                                                          float* __restrict__ out) {
  __shared__ __attribute__((aligned(32))) __bf16 bstage[2][64 * 32];  // [buf][col*32 + k]

  int lane = threadIdx.x & 31;
  int wave = threadIdx.x >> 5;
  long r0 = (long)blockIdx.y * 128 + wave * 16;   // this wave's M-tile
  int  c0 = blockIdx.x * 64;                      // block's 64-column tile
  const __bf16* arow = hs + (r0 + (lane & 15)) * DHID;

  // per-thread 16B slice of the 4 KB stage copy (256 threads * 16B = 4 KB)
  int ccol = threadIdx.x >> 2;                    // 0..63
  int cseg = threadIdx.x & 3;                     // 0..3
  const __bf16* gsrc = WoT + (long)(c0 + ccol) * DHID + cseg * 8;  // + k0 at use site
  __bf16* l0 = &bstage[0][ccol * 32 + cseg * 8];
  __bf16* l1 = &bstage[1][ccol * 32 + cseg * 8];

  // prologue: stage k0 = 0 into buffer 0
  copy16_g2l(l0, gsrc);
  async_copies_done();
  __syncthreads();

  v8f acc[4] = {};
#pragma unroll
  for (int ks = 0; ks < 16; ++ks) {
    int k0 = ks * 32;
    // prefetch next K-stage into the other buffer while we compute
    if (ks < 15) copy16_g2l((ks & 1) ? l0 : l1, gsrc + k0 + 32);

    const __bf16* cbuf = bstage[ks & 1];
    v16bf a = load_a_bf16(arow, k0, lane);
#pragma unroll
    for (int t = 0; t < 4; ++t) {
      const __bf16* lcol = cbuf + (t * 16 + (lane & 15)) * 32 + ((lane >> 4) << 4);
      v16bf bf = *(const v16bf*)lcol;             // 32B from LDS (2x ds_load_b128)
      acc[t] = wmma_bf16(a, bf, acc[t]);
    }
    if (ks < 15) async_copies_done();
    __syncthreads();
  }

  int mbase = 8 * (lane >> 4);
#pragma unroll
  for (int t = 0; t < 4; ++t) {
    int col = c0 + t * 16 + (lane & 15);
    if (col < VOCAB) {
      float bias = bo[col];
#pragma unroll
      for (int i = 0; i < 8; ++i) {
        long row = r0 + mbase + i;
        out[row * VOCAB + col] = acc[t][i] + bias;
      }
    }
  }
}

// ---------------- workspace layout (bytes) ----------------
#define WS_WIT 0L                      // 2048*256*2   = 1,048,576
#define WS_WHT 1048576L                // 2048*512*2   = 2,097,152
#define WS_WOT 3145728L                // 50688*512*2  = 51,904,512
#define WS_XZ  55050240L               // 4096*2048*4  = 33,554,432
#define WS_HS  88604672L               // 4096*512*2   =  4,194,304
#define WS_H0  92798976L               // 16*512*2     =     16,384
#define WS_H1  92815360L
#define WS_BAR 92831744L               // grid-barrier counter

extern "C" void kernel_launch(void* const* d_in, const int* in_sizes, int n_in,
                              void* d_out, int out_size, void* d_ws, size_t ws_size,
                              hipStream_t stream) {
  (void)in_sizes; (void)n_in; (void)out_size; (void)ws_size;
  const int*   idx = (const int*)  d_in[0];
  const float* E   = (const float*)d_in[1];
  const float* Wi  = (const float*)d_in[2];
  const float* Wh  = (const float*)d_in[3];
  const float* bh  = (const float*)d_in[4];
  const float* Wo  = (const float*)d_in[5];
  const float* bo  = (const float*)d_in[6];
  float* out = (float*)d_out;

  char* ws = (char*)d_ws;
  __bf16* WiT = (__bf16*)(ws + WS_WIT);
  __bf16* WhT = (__bf16*)(ws + WS_WHT);
  __bf16* WoT = (__bf16*)(ws + WS_WOT);
  float*  xz  = (float*) (ws + WS_XZ);
  __bf16* hsb = (__bf16*)(ws + WS_HS);
  __bf16* h0  = (__bf16*)(ws + WS_H0);
  __bf16* h1  = (__bf16*)(ws + WS_H1);
  unsigned int* bar = (unsigned int*)(ws + WS_BAR);

  // 0) one-time weight convert+transpose to bf16 (K-major rows for WMMA B-frags)
  { long tot = (long)G4 * DEMB;
    cvt_transpose_kernel<<<(unsigned)((tot + 255) / 256), 256, 0, stream>>>(Wi, WiT, DEMB, G4, G4); }
  { long tot = (long)G4 * DHID;
    cvt_transpose_kernel<<<(unsigned)((tot + 255) / 256), 256, 0, stream>>>(Wh, WhT, DHID, G4, G4); }
  { long tot = (long)VPAD * DHID;
    cvt_transpose_kernel<<<(unsigned)((tot + 255) / 256), 256, 0, stream>>>(Wo, WoT, DHID, VOCAB, VPAD); }
  init_bar_kernel<<<1, 1, 0, stream>>>(bar);

  // 1) embedding gather + input projection (4.3 GFLOP)
  { dim3 g(G4 / 512, SEQ);
    xz_gemm_kernel<<<g, 256, 0, stream>>>(idx, E, WiT, xz); }

  // 2) recurrent scan: 16 WGs = one (intended) workgroup cluster, barrier per step
  lstm_scan_kernel<<<16, 256, 0, stream>>>(xz, WhT, bh, h0, h1, hsb, bar);

  // 3) output head (210.8 GFLOP — the roofline driver)
  { dim3 g(VPAD / 64, ROWS / 128);
    logits_gemm_kernel<<<g, 256, 0, stream>>>(hsb, WoT, bo, out); }
}
